// BitFeedForward2_30631706755717
// MI455X (gfx1250) — 8-bit matrix operand order repaired and checked by reading, compiled, not run on hardware
//
#include <hip/hip_runtime.h>
#include <stdint.h>

typedef __attribute__((ext_vector_type(8))) int v8i;
typedef int b128_t __attribute__((vector_size(16)));

#define HIDDEN 2048
#define INTER  4096
#define TOKENS (4*4096)
#define EPSQ 1e-5f

#if defined(__has_builtin)
#if __has_builtin(__builtin_amdgcn_global_load_async_to_lds_b128) && \
    __has_builtin(__builtin_amdgcn_s_wait_asynccnt)
#define USE_ASYNC_COPY 1
#endif
#endif
#ifndef USE_ASYNC_COPY
#define USE_ASYNC_COPY 0
#endif

#if USE_ASYNC_COPY
#define ASYNC_B128(gsrc, ldst)                                               \
  __builtin_amdgcn_global_load_async_to_lds_b128(                            \
      (__attribute__((address_space(1))) b128_t*)(b128_t*)(void*)(gsrc),     \
      (__attribute__((address_space(3))) b128_t*)(b128_t*)(void*)(ldst), 0, 0)
#endif

__global__ void zero_ws(float* f) { f[0] = 0.f; f[1] = 0.f; }

__global__ void __launch_bounds__(256) absmean_partial(const float* __restrict__ w, long n,
                                                       float* __restrict__ sum) {
  float s = 0.f;
  for (long i = (long)blockIdx.x * blockDim.x + threadIdx.x; i < n;
       i += (long)gridDim.x * blockDim.x)
    s += fabsf(w[i]);
  __shared__ float red[256];
  red[threadIdx.x] = s;
  __syncthreads();
  for (int o = 128; o > 0; o >>= 1) {
    if ((int)threadIdx.x < o) red[threadIdx.x] += red[threadIdx.x + o];
    __syncthreads();
  }
  if (threadIdx.x == 0) atomicAdd(sum, red[0]);
}

__global__ void __launch_bounds__(256) quant_weights(const float* __restrict__ w, long n,
                                                     const float* __restrict__ sum, float invN,
                                                     unsigned* __restrict__ wq,
                                                     float* __restrict__ scale_out) {
  const float scale = fmaxf(sum[0] * invN, EPSQ);
  const long i = (long)blockIdx.x * blockDim.x + threadIdx.x;
  const long nq = n >> 2;
  if (i < nq) {
    unsigned p = 0;
#pragma unroll
    for (int b = 0; b < 4; ++b) {
      float q = rintf(w[i * 4 + b] / scale);
      q = fminf(fmaxf(q, -1.f), 1.f);
      p |= ((unsigned)(((int)q) & 0xFF)) << (8 * b);
    }
    wq[i] = p;
  }
  if (i == 0) scale_out[0] = scale;
}

template <int N>
__global__ void __launch_bounds__(256) fwht_quant(const float* __restrict__ x,
                                                  unsigned* __restrict__ xq,
                                                  float* __restrict__ dscale) {
  __shared__ float buf[N];
  __shared__ float red[256];
  const int tid = threadIdx.x;
  const long base = (long)blockIdx.x * N;

#pragma unroll
  for (int j = 0; j < N / 256; ++j) buf[tid + 256 * j] = x[base + tid + 256 * j];
  __syncthreads();

  for (int h = 1; h < N; h <<= 1) {
    for (int p = tid; p < N / 2; p += 256) {
      const int i = ((p & ~(h - 1)) << 1) | (p & (h - 1));
      const float a = buf[i];
      const float b = buf[i + h];
      buf[i] = a + b;
      buf[i + h] = a - b;
    }
    __syncthreads();
  }

  const float rn = rsqrtf((float)N);

  float m = 0.f;
#pragma unroll
  for (int j = 0; j < N / 256; ++j) m = fmaxf(m, fabsf(buf[tid + 256 * j]));
  red[tid] = m;
  __syncthreads();
  for (int o = 128; o > 0; o >>= 1) {
    if (tid < o) red[tid] = fmaxf(red[tid], red[tid + o]);
    __syncthreads();
  }
  const float maxv = fmaxf(red[0] * rn, EPSQ);
  const float qs = 127.f / maxv;

#pragma unroll
  for (int u = 0; u < N / 1024; ++u) {
    const int ui = tid + 256 * u;
    unsigned p = 0;
#pragma unroll
    for (int b = 0; b < 4; ++b) {
      float q = rintf(buf[ui * 4 + b] * rn * qs);
      q = fminf(fmaxf(q, -127.f), 127.f);
      p |= ((unsigned)(((int)q) & 0xFF)) << (8 * b);
    }
    xq[(base >> 2) + ui] = p;
  }
  if (tid == 0) dscale[blockIdx.x] = maxv / 127.f;
}

template <bool RELU2>
__global__ void __launch_bounds__(256) gemm_iu8(const unsigned* __restrict__ Aq,
                                                const unsigned* __restrict__ Bq,
                                                const float* __restrict__ scaleA,
                                                const float* __restrict__ scaleBp,
                                                float* __restrict__ out, int N, int K) {
  __shared__ unsigned lA[2][128][16];
  __shared__ unsigned lB[2][128][16];

  const int tid = threadIdx.x;
  const int lane = tid & 31;
  const int wave = tid >> 5;
  const int KD = K >> 2;
  const int NK = KD >> 4;
  const long rowA0 = (long)blockIdx.y * 128;
  const long rowB0 = (long)blockIdx.x * 128;

  const int r = tid >> 1;
  const int c = (tid & 1) * 8;
  const unsigned* gA = Aq + (rowA0 + r) * (long)KD + c;
  const unsigned* gB = Bq + (rowB0 + r) * (long)KD + c;

  v8i acc[8] = {};
  const int kh = lane >> 4;
  const int ml = lane & 15;

#if USE_ASYNC_COPY
  ASYNC_B128(gA + 0, &lA[0][r][c]);
  ASYNC_B128(gA + 4, &lA[0][r][c + 4]);
  ASYNC_B128(gB + 0, &lB[0][r][c]);
  ASYNC_B128(gB + 4, &lB[0][r][c + 4]);
  __builtin_amdgcn_s_wait_asynccnt(0);
  __syncthreads();
#else
  uint4 sa0 = *(const uint4*)(gA);
  uint4 sa1 = *(const uint4*)(gA + 4);
  uint4 sb0 = *(const uint4*)(gB);
  uint4 sb1 = *(const uint4*)(gB + 4);
  *(uint4*)&lA[0][r][c] = sa0;
  *(uint4*)&lA[0][r][c + 4] = sa1;
  *(uint4*)&lB[0][r][c] = sb0;
  *(uint4*)&lB[0][r][c + 4] = sb1;
  __syncthreads();
#endif

  for (int kt = 0; kt < NK; ++kt) {
    const int cur = kt & 1;
    const int nxt = cur ^ 1;
    const int koff = (kt + 1) * 16;
    const bool have_next = (kt + 1 < NK);

#if USE_ASYNC_COPY
    if (have_next) {
      ASYNC_B128(gA + koff, &lA[nxt][r][c]);
      ASYNC_B128(gA + koff + 4, &lA[nxt][r][c + 4]);
      ASYNC_B128(gB + koff, &lB[nxt][r][c]);
      ASYNC_B128(gB + koff + 4, &lB[nxt][r][c + 4]);
    }
#else
    uint4 na0, na1, nb0, nb1;
    if (have_next) {
      na0 = *(const uint4*)(gA + koff);
      na1 = *(const uint4*)(gA + koff + 4);
      nb0 = *(const uint4*)(gB + koff);
      nb1 = *(const uint4*)(gB + koff + 4);
      __builtin_prefetch(gA + koff + 16, 0, 0);
      __builtin_prefetch(gB + koff + 16, 0, 0);
    }
#endif

    const unsigned(*cA)[16] = lA[cur];
    const unsigned(*cB)[16] = lB[cur];

    v8i af;
#pragma unroll
    for (int i = 0; i < 8; ++i)
      af[i] = (int)cA[wave * 16 + ml][(i >> 1) * 4 + kh * 2 + (i & 1)];

    v8i bfr[8];
#pragma unroll
    for (int nt = 0; nt < 8; ++nt)
#pragma unroll
      for (int i = 0; i < 8; ++i)
        bfr[nt][i] = (int)cB[nt * 16 + ml][(i >> 1) * 4 + kh * 2 + (i & 1)];

#pragma unroll
    for (int nt = 0; nt < 8; ++nt)
      acc[nt] = __builtin_amdgcn_wmma_i32_16x16x64_iu8(
           true, af,  true, bfr[nt], acc[nt],
           false,  false);

#if USE_ASYNC_COPY
    __builtin_amdgcn_s_wait_asynccnt(0);
    __syncthreads();
#else
    if (have_next) {
      *(uint4*)&lA[nxt][r][c] = na0;
      *(uint4*)&lA[nxt][r][c + 4] = na1;
      *(uint4*)&lB[nxt][r][c] = nb0;
      *(uint4*)&lB[nxt][r][c + 4] = nb1;
    }
    __syncthreads();
#endif
  }

  const float sB = scaleBp[0];
#pragma unroll
  for (int nt = 0; nt < 8; ++nt) {
#pragma unroll
    for (int g = 0; g < 8; ++g) {
      const long row = rowA0 + wave * 16 + g + 8 * kh;
      const long col = rowB0 + nt * 16 + ml;
      float v = (float)acc[nt][g] * scaleA[row] * sB;
      if (RELU2) {
        v = fmaxf(v, 0.f);
        v = v * v;
      }
      out[row * (long)N + col] = v;
    }
  }
}

extern "C" void kernel_launch(void* const* d_in, const int* in_sizes, int n_in,
                              void* d_out, int out_size, void* d_ws, size_t ws_size,
                              hipStream_t stream) {
  (void)in_sizes; (void)n_in; (void)out_size; (void)ws_size;
  const float* x    = (const float*)d_in[0];
  const float* w_up = (const float*)d_in[1];
  const float* w_dn = (const float*)d_in[2];
  float* out = (float*)d_out;

  char* ws = (char*)d_ws;
  float* sums = (float*)ws;
  size_t off = 256;
  float* scales1 = (float*)(ws + off); off += (size_t)TOKENS * 4;
  float* scales2 = (float*)(ws + off); off += (size_t)TOKENS * 4;
  unsigned* X1q  = (unsigned*)(ws + off); off += (size_t)TOKENS * (HIDDEN / 4) * 4;
  unsigned* X2q  = (unsigned*)(ws + off); off += (size_t)TOKENS * (INTER / 4) * 4;
  unsigned* Wupq = (unsigned*)(ws + off); off += (size_t)INTER * (HIDDEN / 4) * 4;
  unsigned* Wdnq = (unsigned*)(ws + off); off += (size_t)HIDDEN * (INTER / 4) * 4;
  float* h       = (float*)(ws + off);    off += (size_t)TOKENS * INTER * 4;

  const long nw = (long)INTER * HIDDEN;
  const float invN = 1.f / (float)nw;

  zero_ws<<<1, 1, 0, stream>>>(sums);
  absmean_partial<<<2048, 256, 0, stream>>>(w_up, nw, sums + 0);
  absmean_partial<<<2048, 256, 0, stream>>>(w_dn, nw, sums + 1);

  const long nw4 = nw >> 2;
  quant_weights<<<(unsigned)((nw4 + 255) / 256), 256, 0, stream>>>(w_up, nw, sums + 0, invN,
                                                                   Wupq, sums + 2);
  quant_weights<<<(unsigned)((nw4 + 255) / 256), 256, 0, stream>>>(w_dn, nw, sums + 1, invN,
                                                                   Wdnq, sums + 3);

  fwht_quant<HIDDEN><<<TOKENS, 256, 0, stream>>>(x, X1q, scales1);

  gemm_iu8<true><<<dim3(INTER / 128, TOKENS / 128), 256, 0, stream>>>(
      X1q, Wupq, scales1, sums + 2, h, INTER, HIDDEN);

  fwht_quant<INTER><<<TOKENS, 256, 0, stream>>>(h, X2q, scales2);

  gemm_iu8<false><<<dim3(HIDDEN / 128, TOKENS / 128), 256, 0, stream>>>(
      X2q, Wdnq, scales2, sums + 3, out, HIDDEN, INTER);
}
